// MultiHeadAttention_85306640433629
// MI455X (gfx1250) — matrix operand order repaired and checked by reading, not compiled, not run on hardware
//
#include <hip/hip_runtime.h>

#define D_MODEL   1024
#define NUM_HEADS 16
#define HEAD_DIM  64
#define SEQ       2048
#define BATCH     2
#define MTOT      (BATCH * SEQ)   // 4096 rows

typedef __attribute__((ext_vector_type(16))) __bf16 v16bf;
typedef __attribute__((ext_vector_type(8)))  float  v8f;
typedef __attribute__((ext_vector_type(4)))  unsigned int v4u;
typedef __attribute__((ext_vector_type(8)))  int    v8i;
typedef __attribute__((ext_vector_type(4)))  int    v4i;

// POD 16-byte vector (avoids HIP vector-type ctor issues inside unions)
struct alignas(16) U4 { unsigned int x, y, z, w; };

union BF16Frag {
    v16bf v;
    U4    q[2];
};

#if __has_builtin(__builtin_amdgcn_tensor_load_to_lds) && \
    __has_builtin(__builtin_amdgcn_s_wait_tensorcnt)
#define USE_TDM 1
#else
#define USE_TDM 0
#endif

__device__ __forceinline__ unsigned short f2bf(float f) {
    unsigned int u = __float_as_uint(f);
    u += 0x7FFFu + ((u >> 16) & 1u);          // round-to-nearest-even
    return (unsigned short)(u >> 16);
}

__device__ __forceinline__ v8f v8f_zero() {
    v8f z;
#pragma unroll
    for (int i = 0; i < 8; ++i) z[i] = 0.0f;
    return z;
}

__device__ __forceinline__ v8f wmma_bf16(const BF16Frag& a, const BF16Frag& b, v8f c) {
    return __builtin_amdgcn_wmma_f32_16x16x32_bf16(
        /*neg_a=*/false, a.v, /*neg_b=*/false, b.v,
        /*c_mod=*/(short)0, c, /*reuse_a=*/false, /*reuse_b=*/false);
}

#if USE_TDM
// byte offset of a __shared__ object within LDS (addrspace(3) view)
__device__ __forceinline__ unsigned lds_off(const void* p) {
    return (unsigned)(size_t)(__attribute__((address_space(3))) const void*)p;
}

// Issue one TDM 2D tile load: rows x cols (bf16) from global row-major tensor
// with row stride 'stride' elements, into LDS at 'ldsaddr'.
__device__ __forceinline__ void tdm_load_tile_bf16(const unsigned short* gptr,
                                                   unsigned ldsaddr,
                                                   int cols, int rows,
                                                   int stride, int tensor_rows) {
    const unsigned long long ga = (unsigned long long)gptr;
    v4u g0;
    g0[0] = 1u;                                     // count=1 (valid), user mode
    g0[1] = ldsaddr;                                // lds_addr (bytes)
    g0[2] = (unsigned)(ga & 0xFFFFFFFFull);         // global_addr[31:0]
    g0[3] = (unsigned)((ga >> 32) & 0x01FFFFFFu)    // global_addr[56:32]
            | (2u << 30);                           // type = 2 ("image")
    v8i g1;
    g1[0] = (int)(1u << 16);                        // data_size = 1 -> 2 bytes
    g1[1] = (int)(((unsigned)cols & 0xFFFFu) << 16);        // tensor_dim0[15:0] @ bits 63:48
    g1[2] = (int)((((unsigned)tensor_rows) & 0xFFFFu) << 16); // tensor_dim1 low16 @ bits 95:80
    g1[3] = (int)(((unsigned)cols & 0xFFFFu) << 16);        // tile_dim0 @ bits 127:112
    g1[4] = (int)((unsigned)rows & 0xFFFFu);                // tile_dim1 @ bits 143:128
    g1[5] = stride;                                 // tensor_dim0_stride low32
    g1[6] = 0;
    g1[7] = 0;
    v4i gz;
    gz[0] = 0; gz[1] = 0; gz[2] = 0; gz[3] = 0;     // groups 2/3 unused (2D)
    v8i gz8;
#pragma unroll
    for (int i = 0; i < 8; ++i) gz8[i] = 0;         // extension group unused
    __builtin_amdgcn_tensor_load_to_lds(g0, g1, gz, gz, gz8, 0);
}
#endif

// ---------------------------------------------------------------------------
// fp32 -> bf16 cast
// ---------------------------------------------------------------------------
__global__ void cast_f32_bf16(const float* __restrict__ in,
                              unsigned short* __restrict__ out, int n) {
    int i = blockIdx.x * blockDim.x + threadIdx.x;
    int stride = gridDim.x * blockDim.x;
    for (; i < n; i += stride) out[i] = f2bf(in[i]);
}

// ---------------------------------------------------------------------------
// C[M,N] = A[M,K] * W[N,K]^T + bias   (bf16 inputs, fp32 accumulate)
// MODE 0: scatter bf16 output into [B, H, S, d] head layout
// MODE 1: fp32 row-major [M, N] output
// Workgroup: 128x128 tile, 8 waves, each wave 32x64 (2x4 WMMA tiles)
// ---------------------------------------------------------------------------
template <int MODE>
__global__ void __launch_bounds__(256)
gemm_bf16_tn(const unsigned short* __restrict__ A,
             const unsigned short* __restrict__ W,
             const float* __restrict__ bias,
             void* __restrict__ out) {
    __shared__ __align__(16) unsigned short As[128 * 32];
    __shared__ __align__(16) unsigned short Bs[128 * 32];

    const int m0   = blockIdx.x * 128;
    const int n0   = blockIdx.y * 128;
    const int tid  = threadIdx.x;
    const int lane = tid & 31;
    const int w    = tid >> 5;
    const int lrow = lane & 15;
    const int hs   = lane >> 4;     // 0 or 1 (half-wave select)
    const int koff = hs * 8;
    const int wm   = (w & 3) * 32;  // 4 waves along M
    const int wn   = (w >> 2) * 64; // 2 waves along N

    v8f acc[2][4];
#pragma unroll
    for (int i = 0; i < 2; ++i)
#pragma unroll
        for (int j = 0; j < 4; ++j) acc[i][j] = v8f_zero();

    const int r  = tid >> 1;
    const int po = (tid & 1) * 16;  // half-row of 16 bf16 (32 bytes)

    for (int kk = 0; kk < D_MODEL; kk += 32) {
        // stage A tile [128 x 32] and W tile [128 x 32] into LDS
        {
            const U4* sa = (const U4*)(A + (size_t)(m0 + r) * D_MODEL + kk + po);
            U4* da = (U4*)(As + r * 32 + po);
            da[0] = sa[0]; da[1] = sa[1];
            const U4* sb = (const U4*)(W + (size_t)(n0 + r) * D_MODEL + kk + po);
            U4* db = (U4*)(Bs + r * 32 + po);
            db[0] = sb[0]; db[1] = sb[1];
        }
        // prefetch next K-step tiles into caches while this tile computes
        if (kk + 32 < D_MODEL) {
            __builtin_prefetch(A + (size_t)(m0 + r) * D_MODEL + kk + 32 + po, 0, 1);
            __builtin_prefetch(W + (size_t)(n0 + r) * D_MODEL + kk + 32 + po, 0, 1);
        }
        __syncthreads();

        BF16Frag af[2], bfr[4];
#pragma unroll
        for (int i = 0; i < 2; ++i) {
            const unsigned short* p = As + (wm + i * 16 + lrow) * 32;
            af[i].q[0] = *(const U4*)(p + hs * 16);
            af[i].q[1] = *(const U4*)(p + hs * 16 + 8);
        }
#pragma unroll
        for (int j = 0; j < 4; ++j) {
            const unsigned short* p = Bs + (wn + j * 16 + lrow) * 32 + hs * 16;
            bfr[j].q[0] = ((const U4*)p)[0];
            bfr[j].q[1] = ((const U4*)p)[1];
        }
#pragma unroll
        for (int i = 0; i < 2; ++i)
#pragma unroll
            for (int j = 0; j < 4; ++j)
                acc[i][j] = wmma_bf16(af[i], bfr[j], acc[i][j]);
        __syncthreads();
    }

    // epilogue: bias + store
#pragma unroll
    for (int i = 0; i < 2; ++i) {
#pragma unroll
        for (int j = 0; j < 4; ++j) {
            const int col = n0 + wn + j * 16 + lrow;
            const float bv = bias[col];
#pragma unroll
            for (int v = 0; v < 8; ++v) {
                const int row = m0 + wm + i * 16 + v + 8 * hs;
                const float val = acc[i][j][v] + bv;
                if (MODE == 0) {
                    const int bb = row >> 11, ss = row & (SEQ - 1);
                    const int hh = col >> 6,  dd = col & 63;
                    ((unsigned short*)out)[(((size_t)(bb * NUM_HEADS + hh) * SEQ + ss) << 6) + dd] =
                        f2bf(val);
                } else {
                    ((float*)out)[(size_t)row * D_MODEL + col] = val;
                }
            }
        }
    }
}

// ---------------------------------------------------------------------------
// Flash-attention: one workgroup per (b, h, 128-query tile).
// 8 waves, each wave owns a 16-query strip; streams 128-key/value tiles.
// K tile staged by the Tensor Data Mover (TDM); V staged transposed by VALU.
// ---------------------------------------------------------------------------
__global__ void __launch_bounds__(256)
mha_attention(const unsigned short* __restrict__ Qh,
              const unsigned short* __restrict__ Kh,
              const unsigned short* __restrict__ Vh,
              unsigned short* __restrict__ ctx) {
    __shared__ __align__(16) unsigned short Ks[128 * 64];      // K tile  [key][d]
    __shared__ __align__(16) unsigned short Vt[64 * 128];      // V^T     [d][key]
    __shared__ __align__(16) unsigned short Ps[8 * 16 * 128];  // per-wave P strip

    const int q0   = blockIdx.x * 128;
    const int bh   = blockIdx.y;         // b * NUM_HEADS + h
    const int b    = bh >> 4;
    const int h    = bh & 15;
    const int tid  = threadIdx.x;
    const int lane = tid & 31;
    const int w    = tid >> 5;
    const int lrow = lane & 15;
    const int hs   = lane >> 4;
    const int koff = hs * 8;

    const size_t headBase = (size_t)bh * SEQ * HEAD_DIM;

    // Q fragments for this wave's 16 rows (kept in registers whole kernel)
    BF16Frag qf[2];
    {
        const unsigned short* qp =
            Qh + headBase + (size_t)(q0 + w * 16 + lrow) * HEAD_DIM;
#pragma unroll
        for (int i = 0; i < 2; ++i) {
            qf[i].q[0] = *(const U4*)(qp + i * 32 + hs * 16);
            qf[i].q[1] = *(const U4*)(qp + i * 32 + hs * 16 + 8);
        }
    }

    float rowm[8], rowl[8];
    v8f oacc[4];
#pragma unroll
    for (int v = 0; v < 8; ++v) { rowm[v] = -1e30f; rowl[v] = 0.0f; }
#pragma unroll
    for (int j = 0; j < 4; ++j) oacc[j] = v8f_zero();

    const float c0 = 0.18033688f;  // (1/sqrt(HEAD_DIM)) * log2(e)

    const int r  = tid >> 1;
    const int po = (tid & 1) * 32;  // half of a 64-elem head row

    for (int kt = 0; kt < SEQ; kt += 128) {
        // ---- stage K tile: TDM DMA (wave 0 issues; TDM ignores EXEC) ----
#if USE_TDM
        if (w == 0) {
            tdm_load_tile_bf16(Kh + headBase + (size_t)kt * HEAD_DIM,
                               lds_off(Ks),
                               /*cols=*/HEAD_DIM, /*rows=*/128,
                               /*stride=*/HEAD_DIM, /*tensor_rows=*/SEQ);
            __builtin_amdgcn_s_wait_tensorcnt(0);
        }
#else
        {
            const unsigned short* kp = Kh + headBase + (size_t)(kt + r) * HEAD_DIM + po;
            U4* kd = (U4*)(Ks + r * 64 + po);
#pragma unroll
            for (int qq = 0; qq < 4; ++qq) kd[qq] = ((const U4*)kp)[qq];
        }
#endif
        // ---- stage V tile transposed into LDS ----
        {
            const unsigned short* vp = Vh + headBase + (size_t)(kt + r) * HEAD_DIM + po;
#pragma unroll
            for (int qq = 0; qq < 4; ++qq) {
                U4 u = ((const U4*)vp)[qq];
                const int d0 = po + qq * 8;
                Vt[(d0 + 0) * 128 + r] = (unsigned short)(u.x);
                Vt[(d0 + 1) * 128 + r] = (unsigned short)(u.x >> 16);
                Vt[(d0 + 2) * 128 + r] = (unsigned short)(u.y);
                Vt[(d0 + 3) * 128 + r] = (unsigned short)(u.y >> 16);
                Vt[(d0 + 4) * 128 + r] = (unsigned short)(u.z);
                Vt[(d0 + 5) * 128 + r] = (unsigned short)(u.z >> 16);
                Vt[(d0 + 6) * 128 + r] = (unsigned short)(u.w);
                Vt[(d0 + 7) * 128 + r] = (unsigned short)(u.w >> 16);
            }
        }
        __syncthreads();

        // ---- scores S = Q K^T : 16 x 128 strip per wave ----
        v8f sacc[8];
#pragma unroll
        for (int t = 0; t < 8; ++t) sacc[t] = v8f_zero();
#pragma unroll
        for (int t = 0; t < 8; ++t) {
#pragma unroll
            for (int hk = 0; hk < 2; ++hk) {
                BF16Frag kf;
                const unsigned short* kp = Ks + (t * 16 + lrow) * 64 + hk * 32 + hs * 16;
                kf.q[0] = ((const U4*)kp)[0];
                kf.q[1] = ((const U4*)kp)[1];
                sacc[t] = wmma_bf16(qf[hk], kf, sacc[t]);
            }
        }

        // ---- online softmax (row stats live in registers, rows align with VGPRs) ----
        float alpha[8];
#pragma unroll
        for (int v = 0; v < 8; ++v) {
            float x = sacc[0][v];
#pragma unroll
            for (int t = 1; t < 8; ++t) x = fmaxf(x, sacc[t][v]);
#pragma unroll
            for (int m = 1; m < 16; m <<= 1) x = fmaxf(x, __shfl_xor(x, m, 32));
            const float mnew = fmaxf(rowm[v], x);
            alpha[v] = exp2f((rowm[v] - mnew) * c0);
            rowm[v]  = mnew;
        }
        float rs[8];
#pragma unroll
        for (int v = 0; v < 8; ++v) rs[v] = 0.0f;
#pragma unroll
        for (int t = 0; t < 8; ++t) {
#pragma unroll
            for (int v = 0; v < 8; ++v) {
                const float p = exp2f((sacc[t][v] - rowm[v]) * c0);
                sacc[t][v] = p;
                rs[v] += p;
            }
        }
#pragma unroll
        for (int v = 0; v < 8; ++v) {
            float x = rs[v];
#pragma unroll
            for (int m = 1; m < 16; m <<= 1) x += __shfl_xor(x, m, 32);
            rowl[v] = rowl[v] * alpha[v] + x;
        }
#pragma unroll
        for (int j = 0; j < 4; ++j)
#pragma unroll
            for (int v = 0; v < 8; ++v) oacc[j][v] *= alpha[v];

        // ---- write P strip to LDS (C-layout -> row-major), reload as A-frags ----
        unsigned short* Pw = Ps + w * (16 * 128);
#pragma unroll
        for (int t = 0; t < 8; ++t)
#pragma unroll
            for (int v = 0; v < 8; ++v)
                Pw[(v + 8 * hs) * 128 + t * 16 + lrow] = f2bf(sacc[t][v]);

        // ---- O += P @ V ----
#pragma unroll
        for (int kb = 0; kb < 4; ++kb) {
            BF16Frag pf;
            const unsigned short* pp = Pw + lrow * 128 + kb * 32;
            pf.q[0] = *(const U4*)(pp + hs * 16);
            pf.q[1] = *(const U4*)(pp + hs * 16 + 8);
#pragma unroll
            for (int j = 0; j < 4; ++j) {
                BF16Frag vf;
                const unsigned short* vpp = Vt + (j * 16 + lrow) * 128 + kb * 32 + hs * 16;
                vf.q[0] = ((const U4*)vpp)[0];
                vf.q[1] = ((const U4*)vpp)[1];
                oacc[j] = wmma_bf16(pf, vf, oacc[j]);
            }
        }
        __syncthreads();
    }

    // ---- normalize and store context [B, S, D_MODEL] (bf16) ----
#pragma unroll
    for (int j = 0; j < 4; ++j) {
#pragma unroll
        for (int v = 0; v < 8; ++v) {
            const int srow = q0 + w * 16 + v + 8 * hs;
            const float o = oacc[j][v] / rowl[v];
            ctx[(size_t)(b * SEQ + srow) * D_MODEL + h * 64 + j * 16 + lrow] = f2bf(o);
        }
    }
}

// ---------------------------------------------------------------------------
// Launcher
// ---------------------------------------------------------------------------
extern "C" void kernel_launch(void* const* d_in, const int* in_sizes, int n_in,
                              void* d_out, int out_size, void* d_ws, size_t ws_size,
                              hipStream_t stream) {
    const float* query = (const float*)d_in[0];
    const float* key   = (const float*)d_in[1];
    const float* value = (const float*)d_in[2];
    const float* Wq    = (const float*)d_in[3];
    const float* bq    = (const float*)d_in[4];
    const float* Wk    = (const float*)d_in[5];
    const float* bk    = (const float*)d_in[6];
    const float* Wv    = (const float*)d_in[7];
    const float* bv    = (const float*)d_in[8];
    const float* Wo    = (const float*)d_in[9];
    const float* bo    = (const float*)d_in[10];

    char* ws = (char*)d_ws;
    const size_t SZ_X = (size_t)MTOT * D_MODEL * 2;     // 8 MB (bf16 activation)
    const size_t SZ_W = (size_t)D_MODEL * D_MODEL * 2;  // 2 MB (bf16 weight)

    unsigned short* Xq  = (unsigned short*)(ws);
    unsigned short* Xk  = (unsigned short*)(ws + SZ_X);
    unsigned short* Xv  = (unsigned short*)(ws + 2 * SZ_X);
    unsigned short* Wqb = (unsigned short*)(ws + 3 * SZ_X);
    unsigned short* Wkb = (unsigned short*)(ws + 3 * SZ_X + 1 * SZ_W);
    unsigned short* Wvb = (unsigned short*)(ws + 3 * SZ_X + 2 * SZ_W);
    unsigned short* Wob = (unsigned short*)(ws + 3 * SZ_X + 3 * SZ_W);
    unsigned short* Qh  = (unsigned short*)(ws + 3 * SZ_X + 4 * SZ_W);
    unsigned short* Kh  = (unsigned short*)(ws + 4 * SZ_X + 4 * SZ_W);
    unsigned short* Vh  = (unsigned short*)(ws + 5 * SZ_X + 4 * SZ_W);
    unsigned short* ctx = (unsigned short*)(ws + 6 * SZ_X + 4 * SZ_W);
    // total workspace: 7*8MB + 4*2MB = 64 MB

    const int nX = MTOT * D_MODEL;     // 4,194,304
    const int nW = D_MODEL * D_MODEL;  // 1,048,576
    cast_f32_bf16<<<1024, 256, 0, stream>>>(query, Xq, nX);
    cast_f32_bf16<<<1024, 256, 0, stream>>>(key,   Xk, nX);
    cast_f32_bf16<<<1024, 256, 0, stream>>>(value, Xv, nX);
    cast_f32_bf16<<<512,  256, 0, stream>>>(Wq, Wqb, nW);
    cast_f32_bf16<<<512,  256, 0, stream>>>(Wk, Wkb, nW);
    cast_f32_bf16<<<512,  256, 0, stream>>>(Wv, Wvb, nW);
    cast_f32_bf16<<<512,  256, 0, stream>>>(Wo, Wob, nW);

    const dim3 ggrid(MTOT / 128, D_MODEL / 128);  // 32 x 8
    gemm_bf16_tn<0><<<ggrid, 256, 0, stream>>>(Xq, Wqb, bq, Qh);
    gemm_bf16_tn<0><<<ggrid, 256, 0, stream>>>(Xk, Wkb, bk, Kh);
    gemm_bf16_tn<0><<<ggrid, 256, 0, stream>>>(Xv, Wvb, bv, Vh);

    mha_attention<<<dim3(SEQ / 128, BATCH * NUM_HEADS), 256, 0, stream>>>(Qh, Kh, Vh, ctx);

    gemm_bf16_tn<1><<<ggrid, 256, 0, stream>>>(ctx, Wob, bo, d_out);
}
